// RelationModule_13477607375708
// MI455X (gfx1250) — matrix operand order repaired and checked by reading, not compiled, not run on hardware
//
#include <hip/hip_runtime.h>
#include <hip/hip_bf16.h>

// ---------------- problem constants ----------------
#define B_IMG   4
#define N_TOK   512
#define C_DIM   1024
#define F_DIM   1024
#define OUT_DIM 1024
#define LOC_DIM 128
#define M_TOTAL (B_IMG * N_TOK)   // 2048

typedef __attribute__((ext_vector_type(16))) __bf16 v16bf;
typedef __attribute__((ext_vector_type(8)))  float  v8f;

__device__ __forceinline__ __bf16 f2bf(float f) { return (__bf16)f; }

// Load one 16-element bf16 fragment as two 16B quads at constant offsets.
// off1 = element distance of the second quad (A and B frags: 8).
__device__ __forceinline__ v16bf load_q2(const __bf16* p, int off1) {
  union { uint4 q[2]; v16bf v; } u;
  u.q[0] = *(const uint4*)(p);
  u.q[1] = *(const uint4*)(p + off1);
  return u.v;
}

__device__ __forceinline__ v8f wmma_bf16(v16bf a, v16bf b, v8f c) {
  return __builtin_amdgcn_wmma_f32_16x16x32_bf16(false, a, false, b,
                                                 (short)0, c, false, false);
}

// ---------------- 32x64 register-blocked, ping-pong double-buffered K loop --
// Fragment register layouts used here (contiguous k, same order in A and B):
//  A (16x32, row-major MxK): lane<16 -> m=lane, K=0..15;
//                            lane>=16 -> m=lane-16, K=16..31.
//  B (32x16 KxN, stored transposed NxK): lane<16 -> n=lane, K=0..15;
//                            lane>=16 -> n=lane-16, K=16..31.
// Per wave: 2 A frags x 4 B frags -> 8 accums, 16 WMMAs per unrolled iter.
// K must be a multiple of 64 (here 1024 or 512).
__device__ __forceinline__ void gemm_tile_32x64(const __bf16* __restrict__ A, int lda,
                                                const __bf16* __restrict__ BT, int ldb,
                                                int mBase, int nBase, int K, int lane,
                                                v8f acc[8]) {
  const __bf16* ap0 = A + (size_t)(mBase + (lane & 15)) * lda + ((lane & 16) ? 16 : 0);
  const __bf16* ap1 = ap0 + (size_t)16 * lda;
  const __bf16* bp0 = BT + (size_t)(nBase + (lane & 15)) * ldb + ((lane & 16) ? 16 : 0);
  const __bf16* bp1 = bp0 + (size_t)16 * ldb;
  const __bf16* bp2 = bp0 + (size_t)32 * ldb;
  const __bf16* bp3 = bp0 + (size_t)48 * ldb;

  v16bf a0[2], b0[4], a1[2], b1[4];
  a0[0] = load_q2(ap0, 8);  a0[1] = load_q2(ap1, 8);
  b0[0] = load_q2(bp0, 8);   b0[1] = load_q2(bp1, 8);
  b0[2] = load_q2(bp2, 8);   b0[3] = load_q2(bp3, 8);

  for (int kb = 0; kb + 64 < K; kb += 64) {
    // prefetch K-step kb+32 into buffer 1
    a1[0] = load_q2(ap0 + 32, 8);  a1[1] = load_q2(ap1 + 32, 8);
    b1[0] = load_q2(bp0 + 32, 8);   b1[1] = load_q2(bp1 + 32, 8);
    b1[2] = load_q2(bp2 + 32, 8);   b1[3] = load_q2(bp3 + 32, 8);
    #pragma unroll
    for (int t = 0; t < 4; ++t) {
      acc[t]     = wmma_bf16(a0[0], b0[t], acc[t]);
      acc[4 + t] = wmma_bf16(a0[1], b0[t], acc[4 + t]);
    }
    ap0 += 64; ap1 += 64; bp0 += 64; bp1 += 64; bp2 += 64; bp3 += 64;
    // prefetch K-step kb+64 into buffer 0
    a0[0] = load_q2(ap0, 8);  a0[1] = load_q2(ap1, 8);
    b0[0] = load_q2(bp0, 8);   b0[1] = load_q2(bp1, 8);
    b0[2] = load_q2(bp2, 8);   b0[3] = load_q2(bp3, 8);
    #pragma unroll
    for (int t = 0; t < 4; ++t) {
      acc[t]     = wmma_bf16(a1[0], b1[t], acc[t]);
      acc[4 + t] = wmma_bf16(a1[1], b1[t], acc[4 + t]);
    }
  }
  // tail: last two K-steps (at current pointers: offsets 0 and 32)
  a1[0] = load_q2(ap0 + 32, 8);  a1[1] = load_q2(ap1 + 32, 8);
  b1[0] = load_q2(bp0 + 32, 8);   b1[1] = load_q2(bp1 + 32, 8);
  b1[2] = load_q2(bp2 + 32, 8);   b1[3] = load_q2(bp3 + 32, 8);
  #pragma unroll
  for (int t = 0; t < 4; ++t) {
    acc[t]     = wmma_bf16(a0[0], b0[t], acc[t]);
    acc[4 + t] = wmma_bf16(a0[1], b0[t], acc[4 + t]);
  }
  #pragma unroll
  for (int t = 0; t < 4; ++t) {
    acc[t]     = wmma_bf16(a1[0], b1[t], acc[t]);
    acc[4 + t] = wmma_bf16(a1[1], b1[t], acc[4 + t]);
  }
}

// ---------------- BN stats: column mean / rstd over M_TOTAL rows ------------
__global__ void bn_stats_kernel(const float* __restrict__ X,
                                float* __restrict__ mean, float* __restrict__ rstd) {
  int col = blockIdx.x * blockDim.x + threadIdx.x;   // 0..C_DIM-1
  float s = 0.f, s2 = 0.f;
  for (int r = 0; r < M_TOTAL; ++r) {
    float v = X[(size_t)r * C_DIM + col];
    s += v; s2 += v * v;
  }
  float m = s * (1.0f / M_TOTAL);
  float var = s2 * (1.0f / M_TOTAL) - m * m;
  mean[col] = m;
  rstd[col] = rsqrtf(var + 1e-5f);
}

__global__ void bn_apply_bf16_kernel(const float* __restrict__ X,
                                     const float* __restrict__ mean,
                                     const float* __restrict__ rstd,
                                     const float* __restrict__ g,
                                     const float* __restrict__ b,
                                     __bf16* __restrict__ Y) {
  int idx = blockIdx.x * blockDim.x + threadIdx.x;
  int col = idx & (C_DIM - 1);
  float v = (X[idx] - mean[col]) * rstd[col] * g[col] + b[col];
  Y[idx] = f2bf(v);
}

__global__ void bn_apply_f32_kernel(const float* __restrict__ X,
                                    const float* __restrict__ mean,
                                    const float* __restrict__ rstd,
                                    const float* __restrict__ g,
                                    const float* __restrict__ b,
                                    float* __restrict__ Y) {
  int idx = blockIdx.x * blockDim.x + threadIdx.x;
  int col = idx & (OUT_DIM - 1);
  Y[idx] = (X[idx] - mean[col]) * rstd[col] * g[col] + b[col];
}

// ---------------- weight transpose + bf16 convert: WT[n*C+k] = W[k*F+n] -----
__global__ void wt_transpose_kernel(const float* __restrict__ W, __bf16* __restrict__ WT) {
  __shared__ float tile[32][33];
  int k0 = blockIdx.x * 32, n0 = blockIdx.y * 32;
  int tx = threadIdx.x & 31, ty = threadIdx.x >> 5;   // 32 x 8
  #pragma unroll
  for (int i = 0; i < 32; i += 8)
    tile[ty + i][tx] = W[(size_t)(k0 + ty + i) * F_DIM + (n0 + tx)];
  __syncthreads();
  #pragma unroll
  for (int i = 0; i < 32; i += 8)
    WT[(size_t)(n0 + ty + i) * C_DIM + (k0 + tx)] = f2bf(tile[tx][ty + i]);
}

// ---------------- q/k/v GEMM: (2048x1024) @ (1024x1024) + bias --------------
// Block = 8 waves as 4(M) x 2(N): block tile 128x128.
// mode 0: store bf16 row-major (q/k). mode 1: store v transposed (B,OUT,N).
__global__ void __launch_bounds__(256)
gemm_qkv_kernel(const __bf16* __restrict__ feat, const __bf16* __restrict__ WT,
                const float* __restrict__ bias,
                __bf16* __restrict__ outQK, __bf16* __restrict__ outVT, int mode) {
  int lane = threadIdx.x & 31;
  int wave = threadIdx.x >> 5;
  int mW = blockIdx.y * 128 + (wave >> 1) * 32;
  int nW = blockIdx.x * 128 + (wave & 1) * 64;
  v8f acc[8] = {};
  gemm_tile_32x64(feat, C_DIM, WT, C_DIM, mW, nW, C_DIM, lane, acc);
  int mLane = mW + ((lane & 16) ? 8 : 0);
  #pragma unroll
  for (int s = 0; s < 2; ++s) {
    #pragma unroll
    for (int t = 0; t < 4; ++t) {
      int n = nW + t * 16 + (lane & 15);
      float bv = bias[n];
      #pragma unroll
      for (int r = 0; r < 8; ++r) {
        int m = mLane + s * 16 + r;
        float val = acc[s * 4 + t][r] + bv;
        if (mode == 0) {
          outQK[(size_t)m * F_DIM + n] = f2bf(val);
        } else {
          int b = m >> 9, mi = m & 511;
          outVT[((size_t)b * OUT_DIM + n) * N_TOK + mi] = f2bf(val);
        }
      }
    }
  }
}

// ---------------- relative positional term rel[b][i][j] ---------------------
__global__ void __launch_bounds__(256)
relpe_kernel(const float* __restrict__ boxes, const float* __restrict__ Wr,
             const float* __restrict__ br, float* __restrict__ rel) {
  int bi = blockIdx.x;            // 0..2047 = b*512 + i
  int b = bi >> 9, i = bi & 511;
  __shared__ float wr[LOC_DIM];
  if (threadIdx.x < LOC_DIM) wr[threadIdx.x] = Wr[threadIdx.x];
  __syncthreads();
  const float* bx = boxes + (size_t)b * N_TOK * 4;
  float xi = bx[i * 4 + 0], yi = bx[i * 4 + 1], wi = bx[i * 4 + 2], hi = bx[i * 4 + 3];
  float cxi = xi + 0.5f * wi, cyi = yi + 0.5f * hi;
  float invpe[16];
  #pragma unroll
  for (int f = 0; f < 16; ++f)
    invpe[f] = __expf(-(2.0f * (float)f / 16.0f) * 6.90775527898f);  // 1/1000^(2f/16)
  float brv = br[0];
  for (int j = threadIdx.x; j < N_TOK; j += 256) {
    float xj = bx[j * 4 + 0], yj = bx[j * 4 + 1], wj = bx[j * 4 + 2], hj = bx[j * 4 + 3];
    float cxj = xj + 0.5f * wj, cyj = yj + 0.5f * hj;
    float l[4];
    l[0] = __logf(fmaxf(fabsf(cxi - cxj) / wj, 1e-3f));
    l[1] = __logf(fmaxf(fabsf(cyi - cyj) / hj, 1e-3f));
    l[2] = __logf(wi / wj);
    l[3] = __logf(hi / hj);
    float s = brv;
    #pragma unroll
    for (int c = 0; c < 4; ++c) {
      #pragma unroll
      for (int f = 0; f < 16; ++f) {
        float a = l[c] * invpe[f];
        float sn, cs; __sincosf(a, &sn, &cs);
        s += sn * wr[c * 32 + f] + cs * wr[c * 32 + 16 + f];
      }
    }
    rel[(size_t)bi * N_TOK + j] = s;
  }
}

// ---------------- scores = q @ k^T * scale + rel  (in-place into rel) -------
// log_softmax(rel) row-constant cancels in the outer softmax, so rel is added
// raw — mathematically identical to the reference.
__global__ void __launch_bounds__(256)
attn_scores_kernel(const __bf16* __restrict__ q, const __bf16* __restrict__ k,
                   float* __restrict__ rel, float scale) {
  int b = blockIdx.z;
  const __bf16* qb = q + (size_t)b * N_TOK * F_DIM;
  const __bf16* kb = k + (size_t)b * N_TOK * F_DIM;  // k is NxK == "BT" for k^T
  float* relb = rel + (size_t)b * N_TOK * N_TOK;
  int lane = threadIdx.x & 31, wave = threadIdx.x >> 5;
  int mW = blockIdx.y * 128 + (wave >> 1) * 32;
  int nW = blockIdx.x * 128 + (wave & 1) * 64;
  v8f acc[8] = {};
  gemm_tile_32x64(qb, F_DIM, kb, F_DIM, mW, nW, F_DIM, lane, acc);
  int mLane = mW + ((lane & 16) ? 8 : 0);
  #pragma unroll
  for (int s = 0; s < 2; ++s) {
    #pragma unroll
    for (int t = 0; t < 4; ++t) {
      int n = nW + t * 16 + (lane & 15);
      #pragma unroll
      for (int r = 0; r < 8; ++r) {
        int m = mLane + s * 16 + r;
        size_t idx = (size_t)m * N_TOK + n;
        relb[idx] = acc[s * 4 + t][r] * scale + relb[idx];
      }
    }
  }
}

// ---------------- row softmax over 512 cols -> bf16 probs -------------------
__global__ void __launch_bounds__(256)
softmax_kernel(const float* __restrict__ scores, __bf16* __restrict__ P) {
  int row = blockIdx.x;                 // 0..2047
  const float* s = scores + (size_t)row * N_TOK;
  __shared__ float red[256];
  int t = threadIdx.x;
  float x0 = s[t], x1 = s[t + 256];
  red[t] = fmaxf(x0, x1);
  __syncthreads();
  for (int o = 128; o > 0; o >>= 1) {
    if (t < o) red[t] = fmaxf(red[t], red[t + o]);
    __syncthreads();
  }
  float mx = red[0];
  __syncthreads();
  float e0 = __expf(x0 - mx), e1 = __expf(x1 - mx);
  red[t] = e0 + e1;
  __syncthreads();
  for (int o = 128; o > 0; o >>= 1) {
    if (t < o) red[t] += red[t + o];
    __syncthreads();
  }
  float inv = 1.0f / red[0];
  P[(size_t)row * N_TOK + t]       = f2bf(e0 * inv);
  P[(size_t)row * N_TOK + t + 256] = f2bf(e1 * inv);
}

// ---------------- out_pre = P @ v  (per image) ------------------------------
__global__ void __launch_bounds__(256)
attn_pv_kernel(const __bf16* __restrict__ P, const __bf16* __restrict__ vT,
               float* __restrict__ outPre) {
  int b = blockIdx.z;
  const __bf16* Pb  = P  + (size_t)b * N_TOK * N_TOK;
  const __bf16* vTb = vT + (size_t)b * OUT_DIM * N_TOK;   // NxK layout
  int lane = threadIdx.x & 31, wave = threadIdx.x >> 5;
  int mW = blockIdx.y * 128 + (wave >> 1) * 32;
  int nW = blockIdx.x * 128 + (wave & 1) * 64;
  v8f acc[8] = {};
  gemm_tile_32x64(Pb, N_TOK, vTb, N_TOK, mW, nW, N_TOK, lane, acc);
  int mLane = mW + ((lane & 16) ? 8 : 0);
  #pragma unroll
  for (int s = 0; s < 2; ++s) {
    #pragma unroll
    for (int t = 0; t < 4; ++t) {
      int n = nW + t * 16 + (lane & 15);
      #pragma unroll
      for (int r = 0; r < 8; ++r) {
        int m = mLane + s * 16 + r;
        outPre[((size_t)b * N_TOK + m) * OUT_DIM + n] = acc[s * 4 + t][r];
      }
    }
  }
}

// ---------------- launcher --------------------------------------------------
extern "C" void kernel_launch(void* const* d_in, const int* in_sizes, int n_in,
                              void* d_out, int out_size, void* d_ws, size_t ws_size,
                              hipStream_t stream) {
  (void)in_sizes; (void)n_in; (void)out_size; (void)ws_size;
  const float* box_features = (const float*)d_in[0];
  const float* boxes        = (const float*)d_in[1];
  const float* bn1_g        = (const float*)d_in[2];
  const float* bn1_b        = (const float*)d_in[3];
  const float* Wq           = (const float*)d_in[4];
  const float* bq           = (const float*)d_in[5];
  const float* Wk           = (const float*)d_in[6];
  const float* bk           = (const float*)d_in[7];
  const float* Wv           = (const float*)d_in[8];
  const float* bv           = (const float*)d_in[9];
  const float* Wr           = (const float*)d_in[10];
  const float* br           = (const float*)d_in[11];
  const float* bn2_g        = (const float*)d_in[12];
  const float* bn2_b        = (const float*)d_in[13];
  float* out = (float*)d_out;

  char* ws = (char*)d_ws;
  size_t off = 0;
  auto alloc = [&](size_t bytes) { char* p = ws + off; off = (off + bytes + 255) & ~(size_t)255; return p; };
  __bf16* featBF = (__bf16*)alloc((size_t)M_TOTAL * C_DIM * 2);          // 4 MB
  __bf16* WqT    = (__bf16*)alloc((size_t)C_DIM * F_DIM * 2);            // 2 MB
  __bf16* WkT    = (__bf16*)alloc((size_t)C_DIM * F_DIM * 2);            // 2 MB
  __bf16* WvT    = (__bf16*)alloc((size_t)C_DIM * OUT_DIM * 2);          // 2 MB
  __bf16* qBF    = (__bf16*)alloc((size_t)M_TOTAL * F_DIM * 2);          // 4 MB
  __bf16* kBF    = (__bf16*)alloc((size_t)M_TOTAL * F_DIM * 2);          // 4 MB
  __bf16* vT     = (__bf16*)alloc((size_t)B_IMG * OUT_DIM * N_TOK * 2);  // 4 MB
  float*  rel    = (float*)alloc((size_t)B_IMG * N_TOK * N_TOK * 4);     // 4 MB
  __bf16* P      = (__bf16*)alloc((size_t)B_IMG * N_TOK * N_TOK * 2);    // 2 MB
  float*  outPre = (float*)alloc((size_t)M_TOTAL * OUT_DIM * 4);         // 8 MB
  float*  mean1  = (float*)alloc(C_DIM * 4);
  float*  rstd1  = (float*)alloc(C_DIM * 4);
  float*  mean2  = (float*)alloc(OUT_DIM * 4);
  float*  rstd2  = (float*)alloc(OUT_DIM * 4);

  // 1) BN1 stats + apply -> bf16 features
  bn_stats_kernel<<<C_DIM / 256, 256, 0, stream>>>(box_features, mean1, rstd1);
  bn_apply_bf16_kernel<<<(M_TOTAL * C_DIM) / 256, 256, 0, stream>>>(
      box_features, mean1, rstd1, bn1_g, bn1_b, featBF);

  // 2) transpose + convert weights to bf16 NxK
  dim3 tg(C_DIM / 32, F_DIM / 32);
  wt_transpose_kernel<<<tg, 256, 0, stream>>>(Wq, WqT);
  wt_transpose_kernel<<<tg, 256, 0, stream>>>(Wk, WkT);
  wt_transpose_kernel<<<tg, 256, 0, stream>>>(Wv, WvT);

  // 3) q, k, v projections (WMMA). M=2048, N=1024 -> grid (8,16), 128x128 tiles
  dim3 gQKV(F_DIM / 128, M_TOTAL / 128);
  gemm_qkv_kernel<<<gQKV, 256, 0, stream>>>(featBF, WqT, bq, qBF, nullptr, 0);
  gemm_qkv_kernel<<<gQKV, 256, 0, stream>>>(featBF, WkT, bk, kBF, nullptr, 0);
  gemm_qkv_kernel<<<gQKV, 256, 0, stream>>>(featBF, WvT, bv, nullptr, vT, 1);

  // 4) relative positional term
  relpe_kernel<<<M_TOTAL, 256, 0, stream>>>(boxes, Wr, br, rel);

  // 5) scores = q k^T / 32 + rel (in-place, WMMA). grid (4,4,4)
  dim3 gS(N_TOK / 128, N_TOK / 128, B_IMG);
  attn_scores_kernel<<<gS, 256, 0, stream>>>(qBF, kBF, rel, 0.03125f);

  // 6) row softmax -> bf16 probs
  softmax_kernel<<<M_TOTAL, 256, 0, stream>>>(rel, P);

  // 7) out_pre = P @ v (WMMA). grid (8,4,4)
  dim3 gPV(OUT_DIM / 128, N_TOK / 128, B_IMG);
  attn_pv_kernel<<<gPV, 256, 0, stream>>>(P, vT, outPre);

  // 8) BN2 stats + apply -> final fp32 output
  bn_stats_kernel<<<OUT_DIM / 256, 256, 0, stream>>>(outPre, mean2, rstd2);
  bn_apply_f32_kernel<<<(M_TOTAL * OUT_DIM) / 256, 256, 0, stream>>>(
      outPre, mean2, rstd2, bn2_g, bn2_b, out);
}
